// NSFFM_76029511073794
// MI455X (gfx1250) — matrix operand order repaired and checked by reading, not compiled, not run on hardware
//
#include <hip/hip_runtime.h>

// ---------------------------------------------------------------------------
// NSFFM on MI455X (gfx1250):
//  - LDS-staged, async-double-buffered bf16 WMMA GEMM (global_load_async_to_lds)
//  - register-blocked direct WMMA GEMM for the small N=64 projection
//  - 3-phase chunked parallel FFA scan (affine-map decomposition)
// ---------------------------------------------------------------------------

typedef __attribute__((ext_vector_type(16))) __bf16 v16bf;
typedef __attribute__((ext_vector_type(8)))  __bf16 bf16x8;
typedef __attribute__((ext_vector_type(8)))  float  v8f;

#define T_LEN 4096
#define D_DIM 512
#define R_DIM 32
#define C_DIM 32
#define L_BLK 2
#define RC_DIM (R_DIM * C_DIM)        // 1024
#define SCALED_DIM (2 * RC_DIM)       // 2048
#define NEG_SLOPE 0.01f
#define LN_EPS 1e-5f

#define NCHUNK 64
#define CHLEN  (T_LEN / NCHUNK)       // 64

// LDS GEMM tiling
#define BM 128
#define BN 64
#define BK 32
#define LDS_STRIDE 40                 // padded row stride (elems): rows 20 banks apart

union Frag16 { v16bf v; bf16x8 h[2]; };

// ---------------------------------------------------------------------------
// CDNA5 async global->LDS copy (ASYNCcnt-tracked).  ISA 15.18.3 opcode 98:
//   global_load_async_to_lds_b128 vdst(LDS byte addr), vaddr(64b), off
// LDS byte offset of a __shared__ object = low 32 bits of its generic address
// (aperture mapping, ISA 10.2).
// ---------------------------------------------------------------------------
__device__ __forceinline__ void async_load_b128(unsigned lds_off, const void* gptr) {
    asm volatile("global_load_async_to_lds_b128 %0, %1, off"
                 :: "v"(lds_off), "v"((unsigned long long)(uintptr_t)gptr)
                 : "memory");
}
__device__ __forceinline__ void wait_async0() {
    asm volatile("s_wait_asynccnt 0x0" ::: "memory");
}

// ---------------------------------------------------------------------------
// f32 -> bf16 conversion (weights / activations staging)
// ---------------------------------------------------------------------------
__global__ void cvt_f32_bf16(const float* __restrict__ in,
                             __bf16* __restrict__ out, int n) {
    int i = blockIdx.x * blockDim.x + threadIdx.x;
    if (i < n) out[i] = (__bf16)in[i];
}

// ---------------------------------------------------------------------------
// Direct register-blocked WMMA GEMM (used for the small N=64 projection).
//   C[M,N] = A[M,K] * W[N,K]^T   (bf16 in, f32 out)
// ---------------------------------------------------------------------------
template <int MT, int NT>
__global__ __launch_bounds__(32) void gemm_wmma_bf16(
    const __bf16* __restrict__ A, const __bf16* __restrict__ W,
    const float* __restrict__ bias, float* __restrict__ Cout,
    int M, int N, int K)
{
    const int n0   = blockIdx.x * (16 * NT);
    const int m0   = blockIdx.y * (16 * MT);
    const int lane = threadIdx.x;
    const int half = lane >> 4;
    const int l16  = lane & 15;

    const __bf16* abase = A + (size_t)(m0 + l16) * K;
    const __bf16* wbase = W + (size_t)(n0 + l16) * K;

    v8f acc[MT][NT];
#pragma unroll
    for (int mi = 0; mi < MT; ++mi)
#pragma unroll
        for (int ni = 0; ni < NT; ++ni) { v8f z = {}; acc[mi][ni] = z; }

    for (int k0 = 0; k0 < K; k0 += 32) {
        Frag16 fa[MT];
#pragma unroll
        for (int mi = 0; mi < MT; ++mi) {
            const __bf16* p = abase + (size_t)mi * 16 * K + k0;
            fa[mi].h[0] = *(const bf16x8*)(p + half * 8);
            fa[mi].h[1] = *(const bf16x8*)(p + 16 + half * 8);
        }
#pragma unroll
        for (int ni = 0; ni < NT; ++ni) {
            const __bf16* p = wbase + (size_t)ni * 16 * K + k0;
            Frag16 fb;
            fb.h[0] = *(const bf16x8*)(p + half * 8);
            fb.h[1] = *(const bf16x8*)(p + 16 + half * 8);
#pragma unroll
            for (int mi = 0; mi < MT; ++mi)
                acc[mi][ni] = __builtin_amdgcn_wmma_f32_16x16x32_bf16(
                    false, fa[mi].v, false, fb.v, (short)0, acc[mi][ni], false, false);
        }
    }

#pragma unroll
    for (int ni = 0; ni < NT; ++ni) {
        const int n = n0 + ni * 16 + l16;
        const float bv = bias ? bias[n] : 0.0f;
#pragma unroll
        for (int mi = 0; mi < MT; ++mi)
#pragma unroll
            for (int v = 0; v < 8; ++v) {
                const int m = m0 + mi * 16 + v + 8 * half;
                Cout[(size_t)m * N + n] = acc[mi][ni][v] + bv;
            }
    }
}

// ---------------------------------------------------------------------------
// LDS-staged async double-buffered WMMA GEMM.
// Workgroup = 4 waves (2x2), macro-tile BM x BN = 128x64, BK=32 per stage.
// Wave (wm,wn) computes a 64x32 sub-tile: MT=4, NT=2 WMMA tiles.
// Per K-step: async-stage next A/B tile into the other LDS buffer while
// computing from the current one; s_wait_asynccnt + barrier at step end.
// ---------------------------------------------------------------------------
__global__ __launch_bounds__(128) void gemm_wmma_lds(
    const __bf16* __restrict__ A,     // [M,K]
    const __bf16* __restrict__ W,     // [N,K]
    const float*  __restrict__ bias,  // [N] or nullptr
    float* __restrict__ Cout,         // [M,N]
    int M, int N, int K)
{
    __shared__ __align__(16) __bf16 As[2][BM * LDS_STRIDE];
    __shared__ __align__(16) __bf16 Bs[2][BN * LDS_STRIDE];

    const int tid  = threadIdx.x;
    const int wave = tid >> 5;
    const int lane = tid & 31;
    const int half = lane >> 4;
    const int l16  = lane & 15;
    const int wm   = wave >> 1;        // 0..1
    const int wn   = wave & 1;         // 0..1
    const int m0   = blockIdx.y * BM;
    const int n0   = blockIdx.x * BN;

    const __bf16* gA = A + (size_t)m0 * K;
    const __bf16* gW = W + (size_t)n0 * K;

    const unsigned ldsA[2] = { (unsigned)(uintptr_t)&As[0][0],
                               (unsigned)(uintptr_t)&As[1][0] };
    const unsigned ldsB[2] = { (unsigned)(uintptr_t)&Bs[0][0],
                               (unsigned)(uintptr_t)&Bs[1][0] };

    // stage one BK-slice into LDS buffer `buf`
    auto stage = [&](int buf, int k0) {
        // A: BM rows x 64B each; thread tid stages row tid (4 x b128)
        {
            const __bf16* src = gA + (size_t)tid * K + k0;
            const unsigned dst = ldsA[buf] + (unsigned)tid * (LDS_STRIDE * 2);
#pragma unroll
            for (int ch = 0; ch < 4; ++ch)
                async_load_b128(dst + ch * 16, src + ch * 8);
        }
        // B: BN rows x 64B; thread tid stages row tid>>1, half-row (2 x b128)
        {
            const int row = tid >> 1;
            const __bf16* src = gW + (size_t)row * K + k0 + (tid & 1) * 16;
            const unsigned dst = ldsB[buf] + (unsigned)row * (LDS_STRIDE * 2)
                               + (unsigned)(tid & 1) * 32;
#pragma unroll
            for (int ch = 0; ch < 2; ++ch)
                async_load_b128(dst + ch * 16, src + ch * 8);
        }
    };

    v8f acc[4][2];
#pragma unroll
    for (int mi = 0; mi < 4; ++mi)
#pragma unroll
        for (int ni = 0; ni < 2; ++ni) { v8f z = {}; acc[mi][ni] = z; }

    const int nk = K / BK;
    stage(0, 0);
    wait_async0();
    __syncthreads();

    for (int kst = 0; kst < nk; ++kst) {
        const int cur = kst & 1;
        if (kst + 1 < nk) stage(cur ^ 1, (kst + 1) * BK);

        const __bf16* aB = &As[cur][0];
        const __bf16* bB = &Bs[cur][0];
        Frag16 fa[4];
#pragma unroll
        for (int mi = 0; mi < 4; ++mi) {
            const __bf16* p = aB + (size_t)(wm * 64 + mi * 16 + l16) * LDS_STRIDE;
            fa[mi].h[0] = *(const bf16x8*)(p + half * 8);
            fa[mi].h[1] = *(const bf16x8*)(p + 16 + half * 8);
        }
#pragma unroll
        for (int ni = 0; ni < 2; ++ni) {
            const __bf16* p = bB + (size_t)(wn * 32 + ni * 16 + l16) * LDS_STRIDE;
            Frag16 fb;
            fb.h[0] = *(const bf16x8*)(p + half * 8);
            fb.h[1] = *(const bf16x8*)(p + 16 + half * 8);
#pragma unroll
            for (int mi = 0; mi < 4; ++mi)
                acc[mi][ni] = __builtin_amdgcn_wmma_f32_16x16x32_bf16(
                    false, fa[mi].v, false, fb.v, (short)0, acc[mi][ni], false, false);
        }

        if (kst + 1 < nk) wait_async0();
        __syncthreads();   // also protects `cur` from being restaged next step
    }

#pragma unroll
    for (int ni = 0; ni < 2; ++ni) {
        const int n = n0 + wn * 32 + ni * 16 + l16;
        const float bv = bias ? bias[n] : 0.0f;
#pragma unroll
        for (int mi = 0; mi < 4; ++mi)
#pragma unroll
            for (int v = 0; v < 8; ++v) {
                const int m = m0 + wm * 64 + mi * 16 + v + 8 * half;
                Cout[(size_t)m * N + n] = acc[mi][ni][v] + bv;
            }
    }
}

// ---------------------------------------------------------------------------
// denomInv[t] = 1 / (1e-8 + (sum_r |tr[t,r]|) * (sum_c |ct[t,c]|))
// ---------------------------------------------------------------------------
__global__ void rowsum_kernel(const float* __restrict__ trct,
                              float* __restrict__ denomInv, int T) {
    int t = blockIdx.x * blockDim.x + threadIdx.x;
    if (t >= T) return;
    const float* row = trct + (size_t)t * 64;
    float s1 = 0.f, s2 = 0.f;
#pragma unroll 8
    for (int j = 0; j < 32; ++j) {
        s1 += fabsf(row[j]);
        s2 += fabsf(row[32 + j]);
    }
    denomInv[t] = 1.0f / (1e-8f + s1 * s2);
}

// ---------------------------------------------------------------------------
// FFA scan: s_t = pre_t + gamma*s_{t-1}, reset on start[t].
// Chunk map is affine: s_out = B_k + A_k*s_in, A_k = resetAny ? 0 : gamma^CHLEN.
// ---------------------------------------------------------------------------
__device__ __forceinline__ float ffa_pre(const float* __restrict__ trct,
                                         const float* __restrict__ denomInv,
                                         int t, int r, int c) {
    const float trv = trct[(size_t)t * 64 + r];
    const float ctv = trct[(size_t)t * 64 + 32 + c];
    return fabsf(trv * ctv) * denomInv[t];
}

__device__ __forceinline__ void ffa_step(float pre, int st, float gre, float gim,
                                         float& sre, float& sim) {
    if (st) { sre = pre; sim = 0.0f; }
    else {
        const float nre = pre + gre * sre - gim * sim;
        const float nim = gre * sim + gim * sre;
        sre = nre; sim = nim;
    }
}

__global__ __launch_bounds__(1024) void ffa_scan_phase1(
    const float* __restrict__ trct, const float* __restrict__ denomInv,
    const int* __restrict__ start,
    const float* __restrict__ a, const float* __restrict__ bfreq,
    float* __restrict__ chunkRe, float* __restrict__ chunkIm,
    int* __restrict__ chunkReset)
{
    const int k = blockIdx.x;
    const int tid = threadIdx.x;
    const int r = tid >> 5;
    const int c = tid & 31;
    const float dec = __expf(-fabsf(a[r]));
    const float gre = dec * __cosf(bfreq[c]);
    const float gim = dec * __sinf(bfreq[c]);

    float sre = 0.f, sim = 0.f;
    const int t0 = k * CHLEN;
    for (int tt = 0; tt < CHLEN; ++tt) {
        const int t = t0 + tt;
        ffa_step(ffa_pre(trct, denomInv, t, r, c), start[t], gre, gim, sre, sim);
    }
    chunkRe[(size_t)k * RC_DIM + tid] = sre;
    chunkIm[(size_t)k * RC_DIM + tid] = sim;
    if (tid == 0) {
        int rs = 0;
        for (int tt = 0; tt < CHLEN; ++tt) rs |= start[t0 + tt];
        chunkReset[k] = rs;
    }
}

__global__ __launch_bounds__(1024) void ffa_scan_phase2(
    const float* __restrict__ chunkRe, const float* __restrict__ chunkIm,
    const int* __restrict__ chunkReset,
    const float* __restrict__ a, const float* __restrict__ bfreq,
    const float* __restrict__ s0re, const float* __restrict__ s0im,
    float* __restrict__ carryRe, float* __restrict__ carryIm)
{
    const int tid = threadIdx.x;
    const int r = tid >> 5;
    const int c = tid & 31;
    const float decC = __expf(-fabsf(a[r]) * (float)CHLEN);
    const float ang  = bfreq[c] * (float)CHLEN;
    const float gCre = decC * __cosf(ang);
    const float gCim = decC * __sinf(ang);

    float cre = s0re[tid], cim = s0im[tid];
    for (int k = 0; k < NCHUNK; ++k) {
        carryRe[(size_t)k * RC_DIM + tid] = cre;
        carryIm[(size_t)k * RC_DIM + tid] = cim;
        const float fre = chunkRe[(size_t)k * RC_DIM + tid];
        const float fim = chunkIm[(size_t)k * RC_DIM + tid];
        if (chunkReset[k]) { cre = fre; cim = fim; }
        else {
            const float nre = fre + gCre * cre - gCim * cim;
            const float nim = fim + gCre * cim + gCim * cre;
            cre = nre; cim = nim;
        }
    }
}

__global__ __launch_bounds__(1024) void ffa_scan_phase3(
    const float* __restrict__ trct, const float* __restrict__ denomInv,
    const int* __restrict__ start,
    const float* __restrict__ a, const float* __restrict__ bfreq,
    const float* __restrict__ carryRe, const float* __restrict__ carryIm,
    __bf16* __restrict__ scaledbf)
{
    const int k = blockIdx.x;
    const int tid = threadIdx.x;
    const int r = tid >> 5;
    const int c = tid & 31;
    const float dec = __expf(-fabsf(a[r]));
    const float gre = dec * __cosf(bfreq[c]);
    const float gim = dec * __sinf(bfreq[c]);

    float sre = carryRe[(size_t)k * RC_DIM + tid];
    float sim = carryIm[(size_t)k * RC_DIM + tid];
    const int t0 = k * CHLEN;
    for (int tt = 0; tt < CHLEN; ++tt) {
        const int t = t0 + tt;
        ffa_step(ffa_pre(trct, denomInv, t, r, c), start[t], gre, gim, sre, sim);
        const float mag = sqrtf(sre * sre + sim * sim);
        const float lg  = log1pf(mag);
        const float s   = (mag > 0.0f) ? (lg / mag) : 0.0f;
        scaledbf[(size_t)t * SCALED_DIM + tid]          = (__bf16)(s * sim);
        scaledbf[(size_t)t * SCALED_DIM + RC_DIM + tid] = (__bf16)(s * sre);
    }
}

// ---------------------------------------------------------------------------
// Row LayerNorm + leaky-ReLU, optional bf16 mirror and residual h += z.
// ---------------------------------------------------------------------------
__global__ __launch_bounds__(256) void ln_lrelu_kernel(
    const float* __restrict__ y, const float* __restrict__ g,
    const float* __restrict__ beta,
    float* __restrict__ zout, __bf16* __restrict__ zbf,
    float* __restrict__ hresid, __bf16* __restrict__ hbf, int D)
{
    const int t = blockIdx.x;
    const float* yrow = y + (size_t)t * D;
    __shared__ float red0[8];
    __shared__ float red1[8];

    float s = 0.f, s2 = 0.f;
    for (int j = threadIdx.x; j < D; j += blockDim.x) {
        const float v = yrow[j];
        s += v; s2 += v * v;
    }
#pragma unroll
    for (int off = 16; off > 0; off >>= 1) {
        s  += __shfl_down(s,  off, 32);
        s2 += __shfl_down(s2, off, 32);
    }
    const int wid  = threadIdx.x >> 5;
    const int lane = threadIdx.x & 31;
    if (lane == 0) { red0[wid] = s; red1[wid] = s2; }
    __syncthreads();
    if (wid == 0) {
        s  = (lane < 8) ? red0[lane] : 0.f;
        s2 = (lane < 8) ? red1[lane] : 0.f;
#pragma unroll
        for (int off = 4; off > 0; off >>= 1) {
            s  += __shfl_down(s,  off, 32);
            s2 += __shfl_down(s2, off, 32);
        }
        if (lane == 0) { red0[0] = s; red1[0] = s2; }
    }
    __syncthreads();

    const float invD = 1.0f / (float)D;
    const float mean = red0[0] * invD;
    const float var  = red1[0] * invD - mean * mean;
    const float rstd = rsqrtf(var + LN_EPS);

    for (int j = threadIdx.x; j < D; j += blockDim.x) {
        float v = (yrow[j] - mean) * rstd * g[j] + beta[j];
        v = (v > 0.0f) ? v : NEG_SLOPE * v;
        const size_t idx = (size_t)t * D + j;
        zout[idx] = v;
        if (zbf) zbf[idx] = (__bf16)v;
        if (hresid) {
            const float hn = hresid[idx] + v;
            hresid[idx] = hn;
            if (hbf) hbf[idx] = (__bf16)hn;
        }
    }
}

// ---------------------------------------------------------------------------
// Host driver
// ---------------------------------------------------------------------------
static inline size_t align256(size_t x) { return (x + 255) & ~(size_t)255; }

extern "C" void kernel_launch(void* const* d_in, const int* in_sizes, int n_in,
                              void* d_out, int out_size, void* d_ws, size_t ws_size,
                              hipStream_t stream) {
    (void)in_sizes; (void)n_in; (void)out_size; (void)ws_size;

    const float* x          = (const float*)d_in[0];
    const int*   start      = (const int*)  d_in[1];
    /* d_in[2] next_done: unused by the math */
    const float* state_real = (const float*)d_in[3];
    const float* state_imag = (const float*)d_in[4];
    const float* Wt         = (const float*)d_in[5];
    const float* Wc         = (const float*)d_in[6];
    const float* a          = (const float*)d_in[7];
    const float* b          = (const float*)d_in[8];
    const float* W0         = (const float*)d_in[9];
    const float* b0         = (const float*)d_in[10];
    const float* g0         = (const float*)d_in[11];
    const float* beta0      = (const float*)d_in[12];
    const float* W1         = (const float*)d_in[13];
    const float* b1         = (const float*)d_in[14];
    const float* g1         = (const float*)d_in[15];
    const float* beta1      = (const float*)d_in[16];

    char* ws = (char*)d_ws;
    size_t off = 0;
    auto carve = [&](size_t bytes) -> void* {
        void* p = ws + off;
        off = align256(off + bytes);
        return p;
    };
    float*  h        = (float*) carve((size_t)T_LEN * D_DIM * 4);
    __bf16* hbf      = (__bf16*)carve((size_t)T_LEN * D_DIM * 2);
    float*  trct     = (float*) carve((size_t)T_LEN * 64 * 4);
    float*  denomInv = (float*) carve((size_t)T_LEN * 4);
    __bf16* scaledbf = (__bf16*)carve((size_t)T_LEN * SCALED_DIM * 2);
    float*  ybuf     = (float*) carve((size_t)T_LEN * D_DIM * 4);
    float*  z0       = (float*) carve((size_t)T_LEN * D_DIM * 4);
    __bf16* z0bf     = (__bf16*)carve((size_t)T_LEN * D_DIM * 2);
    __bf16* wtrctbf  = (__bf16*)carve((size_t)64 * D_DIM * 2);
    __bf16* w0bf     = (__bf16*)carve((size_t)D_DIM * SCALED_DIM * 2);
    __bf16* w1bf     = (__bf16*)carve((size_t)D_DIM * D_DIM * 2);
    float*  chunkRe  = (float*) carve((size_t)NCHUNK * RC_DIM * 4);
    float*  chunkIm  = (float*) carve((size_t)NCHUNK * RC_DIM * 4);
    float*  carryRe  = (float*) carve((size_t)NCHUNK * RC_DIM * 4);
    float*  carryIm  = (float*) carve((size_t)NCHUNK * RC_DIM * 4);
    int*    chunkRst = (int*)   carve((size_t)NCHUNK * 4);
    float*  zout     = (float*)d_out;

    const int CVT = 256;

    hipMemcpyAsync(h, x, (size_t)T_LEN * D_DIM * 4, hipMemcpyDeviceToDevice, stream);
    cvt_f32_bf16<<<(T_LEN * D_DIM + CVT - 1) / CVT, CVT, 0, stream>>>(h, hbf, T_LEN * D_DIM);

    for (int i = 0; i < L_BLK; ++i) {
        const bool last = (i == L_BLK - 1);
        const float* ai = a + (size_t)i * R_DIM;
        const float* bi = b + (size_t)i * C_DIM;

        // --- trace/context projections: trct[T,64] = hbf @ [Wt;Wc]^T ---
        cvt_f32_bf16<<<(R_DIM * D_DIM + CVT - 1) / CVT, CVT, 0, stream>>>(
            Wt + (size_t)i * R_DIM * D_DIM, wtrctbf, R_DIM * D_DIM);
        cvt_f32_bf16<<<(R_DIM * D_DIM + CVT - 1) / CVT, CVT, 0, stream>>>(
            Wc + (size_t)i * R_DIM * D_DIM, wtrctbf + (size_t)R_DIM * D_DIM, R_DIM * D_DIM);
        gemm_wmma_bf16<1, 4><<<dim3(1, T_LEN / 16), 32, 0, stream>>>(
            hbf, wtrctbf, nullptr, trct, T_LEN, 64, D_DIM);

        // --- factorized outer-product normalizer ---
        rowsum_kernel<<<T_LEN / 256, 256, 0, stream>>>(trct, denomInv, T_LEN);

        // --- 3-phase chunked parallel scan ---
        ffa_scan_phase1<<<NCHUNK, 1024, 0, stream>>>(
            trct, denomInv, start, ai, bi, chunkRe, chunkIm, chunkRst);
        ffa_scan_phase2<<<1, 1024, 0, stream>>>(
            chunkRe, chunkIm, chunkRst, ai, bi,
            state_real + (size_t)i * RC_DIM, state_imag + (size_t)i * RC_DIM,
            carryRe, carryIm);
        ffa_scan_phase3<<<NCHUNK, 1024, 0, stream>>>(
            trct, denomInv, start, ai, bi, carryRe, carryIm, scaledbf);

        // --- z = lrelu(LN(scaled @ W0^T + b0)) ---
        cvt_f32_bf16<<<(D_DIM * SCALED_DIM + CVT - 1) / CVT, CVT, 0, stream>>>(
            W0 + (size_t)i * D_DIM * SCALED_DIM, w0bf, D_DIM * SCALED_DIM);
        gemm_wmma_lds<<<dim3(D_DIM / BN, T_LEN / BM), 128, 0, stream>>>(
            scaledbf, w0bf, b0 + (size_t)i * D_DIM, ybuf, T_LEN, D_DIM, SCALED_DIM);
        ln_lrelu_kernel<<<T_LEN, 256, 0, stream>>>(
            ybuf, g0 + (size_t)i * D_DIM, beta0 + (size_t)i * D_DIM,
            z0, z0bf, nullptr, nullptr, D_DIM);

        // --- z = lrelu(LN(z @ W1^T + b1)); fuse residual for next block ---
        cvt_f32_bf16<<<(D_DIM * D_DIM + CVT - 1) / CVT, CVT, 0, stream>>>(
            W1 + (size_t)i * D_DIM * D_DIM, w1bf, D_DIM * D_DIM);
        gemm_wmma_lds<<<dim3(D_DIM / BN, T_LEN / BM), 128, 0, stream>>>(
            z0bf, w1bf, b1 + (size_t)i * D_DIM, ybuf, T_LEN, D_DIM, D_DIM);
        ln_lrelu_kernel<<<T_LEN, 256, 0, stream>>>(
            ybuf, g1 + (size_t)i * D_DIM, beta1 + (size_t)i * D_DIM,
            zout, (__bf16*)nullptr,
            last ? nullptr : h, last ? nullptr : hbf, D_DIM);
    }
}
